// GCN_24318104830572
// MI455X (gfx1250) — matrix operand order repaired and checked by reading, not compiled, not run on hardware
//
#include <hip/hip_runtime.h>
#include <hip/hip_bf16.h>

typedef __attribute__((ext_vector_type(4)))  float  v4f;
typedef __attribute__((ext_vector_type(8)))  float  v8f;
typedef __attribute__((ext_vector_type(8)))  __bf16 v8bf;
typedef __attribute__((ext_vector_type(16))) __bf16 v16bf;

#define NNODE 8192
#define HID   200
#define HIDP  208      // 200 padded to 13 * 16
#define NT    13       // number of 16-wide n-tiles in HIDP
#define OUTC  5
#define KC    256      // K chunk staged in LDS
#define KSTR  264      // padded LDS row length in bf16 (528 B: 16B-aligned, bank-skewed)

// ---- A fragment: 16x32 bf16 from a row-major f32 matrix (ld = NNODE) ----
// Layout (contiguous k order, matches B frags): lanes 0-15 hold K 0-15,
// lanes 16-31 hold K 16-31.
static __device__ __forceinline__ v16bf load_a_frag_f32(
    const float* __restrict__ A, int row, int k0, int lane) {
  const int kb = k0 + ((lane >> 4) << 4);
  const float* p = A + (size_t)row * NNODE + kb;
  v4f f0 = *(const v4f*)(p + 0);
  v4f f1 = *(const v4f*)(p + 4);
  v4f f2 = *(const v4f*)(p + 8);
  v4f f3 = *(const v4f*)(p + 12);
  v16bf a;
#pragma unroll
  for (int i = 0; i < 4; ++i) {
    a[i + 0]  = (__bf16)f0[i];
    a[i + 4]  = (__bf16)f1[i];
    a[i + 8]  = (__bf16)f2[i];
    a[i + 12] = (__bf16)f3[i];
  }
  return a;
}

static __device__ __forceinline__ v16bf combine16(v8bf lo, v8bf hi) {
  v16bf b;
#pragma unroll
  for (int i = 0; i < 8; ++i) { b[i] = lo[i]; b[i + 8] = hi[i]; }
  return b;
}

static __device__ __forceinline__ v16bf load_b_frag(const __bf16* bp, int kk) {
  return combine16(*(const v8bf*)(bp + kk), *(const v8bf*)(bp + kk + 8));
}

// Compile-time unrolled 13-WMMA burst: J is a template parameter so the
// reuse_a hint is an integer constant expression; B frags are software-
// pipelined (load J+1 before WMMA J) to avoid WAR-hazard NOPs and exposed
// LDS latency.
template <int J, int NTT>
struct MMTiles {
  static __device__ __forceinline__ void run(
      const v16bf& a, const __bf16* const (&bp)[NTT], v8f (&acc)[NTT],
      int kk, v16bf bcur) {
    v16bf bnxt = bcur;
    if constexpr (J + 1 < NTT) bnxt = load_b_frag(bp[J + 1], kk);
    acc[J] = __builtin_amdgcn_wmma_f32_16x16x32_bf16(
        false, a, false, bcur, (short)0, acc[J],
        /*reuse_a=*/(J + 1 < NTT), /*reuse_b=*/false);
    if constexpr (J + 1 < NTT)
      MMTiles<J + 1, NTT>::run(a, bp, acc, kk, bnxt);
  }
};

// ================= Kernel 1: S1t[n][m] = bf16( (x @ W1)[m][n] ) =============
__global__ void __launch_bounds__(128)
k_gemm_xw1(const float* __restrict__ x, const float* __restrict__ W1,
           __bf16* __restrict__ S1t) {
  extern __shared__ __bf16 Bl[];               // [HIDP][KSTR] bf16
  const int lane  = threadIdx.x & 31;
  const int wave  = threadIdx.x >> 5;
  const int m0    = (blockIdx.x * 4 + wave) << 4;
  const int arow  = m0 + (lane & 15);
  const int nlo   = lane & 15;
  const int khalf = (lane >> 4) << 4;

  // Per-tile LDS base pointers (addrspace(3) => 1 VGPR each); DS offsets
  // inside the k-loop become small immediates after full unroll.
  const __bf16* bp[NT];
#pragma unroll
  for (int j = 0; j < NT; ++j) bp[j] = Bl + (j * 16 + nlo) * KSTR + khalf;

  v8f acc[NT];
  const v8f vzero = {0.f,0.f,0.f,0.f,0.f,0.f,0.f,0.f};
#pragma unroll
  for (int j = 0; j < NT; ++j) acc[j] = vzero;

  for (int kc = 0; kc < NNODE; kc += KC) {
    __syncthreads();
    // Stage W1 chunk: transpose [k][n] -> [n][k], f32->bf16, zero-pad n>=200
    for (int idx = threadIdx.x; idx < KC * HIDP; idx += 128) {
      const int n = idx % HIDP;
      const int k = idx / HIDP;
      const float v = (n < HID) ? W1[(size_t)(kc + k) * HID + n] : 0.f;
      Bl[n * KSTR + k] = (__bf16)v;
    }
    __syncthreads();

    if (kc + KC < NNODE) {  // prefetch next A chunk (2 lines/lane covers 256 f32)
      const float* pfp = x + (size_t)arow * NNODE + kc + KC + ((lane >> 4) << 7);
      __builtin_prefetch(pfp, 0, 3);
      __builtin_prefetch(pfp + 64, 0, 3);
    }

#pragma unroll
    for (int kk = 0; kk < KC; kk += 32) {
      v16bf a = load_a_frag_f32(x, arow, kc + kk, lane);
      v16bf b0 = load_b_frag(bp[0], kk);
      MMTiles<0, NT>::run(a, bp, acc, kk, b0);
    }
  }

  const int mbase = m0 + ((lane >> 4) << 3);
#pragma unroll
  for (int j = 0; j < NT; ++j) {
    const int n = j * 16 + nlo;
#pragma unroll
    for (int r = 0; r < 8; ++r)
      S1t[(size_t)n * NNODE + mbase + r] = (__bf16)acc[j][r];
  }
}

// ====== Kernel 2: H[m][n] = bf16(relu( (adj @ S1)[m][n] + b1[n] )) ==========
__global__ void __launch_bounds__(128)
k_gemm_adj_s1(const float* __restrict__ adj, const __bf16* __restrict__ S1t,
              const float* __restrict__ b1, __bf16* __restrict__ H) {
  extern __shared__ __bf16 Bl[];               // [HIDP][KSTR] bf16
  const int lane  = threadIdx.x & 31;
  const int wave  = threadIdx.x >> 5;
  const int m0    = (blockIdx.x * 4 + wave) << 4;
  const int arow  = m0 + (lane & 15);
  const int nlo   = lane & 15;
  const int khalf = (lane >> 4) << 4;

  const __bf16* bp[NT];
#pragma unroll
  for (int j = 0; j < NT; ++j) bp[j] = Bl + (j * 16 + nlo) * KSTR + khalf;

  v8f acc[NT];
  const v8f vzero = {0.f,0.f,0.f,0.f,0.f,0.f,0.f,0.f};
#pragma unroll
  for (int j = 0; j < NT; ++j) acc[j] = vzero;

  for (int kc = 0; kc < NNODE; kc += KC) {
    __syncthreads();
    // S1t is already bf16 [n][k]: linear copy -> async global->LDS (ASYNCcnt),
    // bypassing VGPRs entirely.
    for (int idx = threadIdx.x; idx < HIDP * (KC / 8); idx += 128) {
      const int n = idx >> 5;            // KC/8 == 32
      const int o = (idx & 31) << 3;
      unsigned lds_off = (unsigned)(unsigned long long)(Bl + n * KSTR + o);
      const __bf16* gaddr = S1t + (size_t)n * NNODE + kc + o;
      asm volatile("global_load_async_to_lds_b128 %0, %1, off"
                   :: "v"(lds_off), "v"(gaddr) : "memory");
    }
    asm volatile("s_wait_asynccnt 0x0" ::: "memory");
    __syncthreads();

    if (kc + KC < NNODE) {
      const float* pfp = adj + (size_t)arow * NNODE + kc + KC + ((lane >> 4) << 7);
      __builtin_prefetch(pfp, 0, 3);
      __builtin_prefetch(pfp + 64, 0, 3);
    }

#pragma unroll
    for (int kk = 0; kk < KC; kk += 32) {
      v16bf a = load_a_frag_f32(adj, arow, kc + kk, lane);
      v16bf b0 = load_b_frag(bp[0], kk);
      MMTiles<0, NT>::run(a, bp, acc, kk, b0);
    }
  }

  const int mbase = m0 + ((lane >> 4) << 3);
#pragma unroll
  for (int j = 0; j < NT; ++j) {
    const int n = j * 16 + nlo;
    const float bb = (n < HID) ? b1[n] : 0.f;
#pragma unroll
    for (int r = 0; r < 8; ++r) {
      float h = acc[j][r] + bb;
      h = h > 0.f ? h : 0.f;
      H[(size_t)(mbase + r) * HIDP + n] = (__bf16)h;
    }
  }
}

// ======= Kernel 2b: S2t[c][m] = bf16( (H @ W2)[m][c] ), rows 5..15 = 0 ======
__global__ void __launch_bounds__(256)
k_h_w2(const __bf16* __restrict__ H, const float* __restrict__ W2,
       __bf16* __restrict__ S2t) {
  __shared__ float w2s[HID * OUTC];
  for (int i = threadIdx.x; i < HID * OUTC; i += 256) w2s[i] = W2[i];
  __syncthreads();
  const int m = blockIdx.x * 256 + threadIdx.x;
  float s0 = 0.f, s1 = 0.f, s2 = 0.f, s3 = 0.f, s4 = 0.f;
  for (int n8 = 0; n8 < HID; n8 += 8) {
    v8bf hv = *(const v8bf*)(H + (size_t)m * HIDP + n8);
#pragma unroll
    for (int i = 0; i < 8; ++i) {
      const float h = (float)hv[i];
      const float* w = &w2s[(n8 + i) * OUTC];
      s0 += h * w[0]; s1 += h * w[1]; s2 += h * w[2];
      s3 += h * w[3]; s4 += h * w[4];
    }
  }
  S2t[(size_t)0 * NNODE + m] = (__bf16)s0;
  S2t[(size_t)1 * NNODE + m] = (__bf16)s1;
  S2t[(size_t)2 * NNODE + m] = (__bf16)s2;
  S2t[(size_t)3 * NNODE + m] = (__bf16)s3;
  S2t[(size_t)4 * NNODE + m] = (__bf16)s4;
#pragma unroll
  for (int c = OUTC; c < 16; ++c)
    S2t[(size_t)c * NNODE + m] = (__bf16)0.f;
}

// ==== Kernel 3: out = softmax( adj @ S2 + b2 ), one 16x16 WMMA tile/wave ====
__global__ void __launch_bounds__(128)
k_gemm_adj_s2_softmax(const float* __restrict__ adj,
                      const __bf16* __restrict__ S2t,
                      const float* __restrict__ b2, float* __restrict__ out) {
  __shared__ float Ls[4][16 * 16];
  const int lane  = threadIdx.x & 31;
  const int wave  = threadIdx.x >> 5;
  const int m0    = (blockIdx.x * 4 + wave) << 4;
  const int arow  = m0 + (lane & 15);
  const int nlo   = lane & 15;
  const int khalf = (lane >> 4) << 4;

  v8f acc = {0.f,0.f,0.f,0.f,0.f,0.f,0.f,0.f};
  const __bf16* bbase = S2t + (size_t)nlo * NNODE + khalf;   // L2-resident B
#pragma unroll 4
  for (int k0 = 0; k0 < NNODE; k0 += 32) {
    if ((k0 & 255) == 0 && k0 + 1024 < NNODE)
      __builtin_prefetch(adj + (size_t)arow * NNODE + k0 + 1024 + ((lane >> 4) << 7), 0, 3);
    v16bf a = load_a_frag_f32(adj, arow, k0, lane);
    v16bf b = combine16(*(const v8bf*)(bbase + k0), *(const v8bf*)(bbase + k0 + 8));
    acc = __builtin_amdgcn_wmma_f32_16x16x32_bf16(
        false, a, false, b, (short)0, acc, false, false);
  }

  const int mloc = (lane >> 4) << 3;
#pragma unroll
  for (int r = 0; r < 8; ++r)
    Ls[wave][(mloc + r) * 16 + nlo] = acc[r];
  __syncthreads();

  if (lane < 16) {
    const int m = m0 + lane;
    float l[OUTC];
    float mx = -3.0e38f;
#pragma unroll
    for (int c = 0; c < OUTC; ++c) {
      l[c] = Ls[wave][lane * 16 + c] + b2[c];
      mx = fmaxf(mx, l[c]);
    }
    float se = 0.f;
#pragma unroll
    for (int c = 0; c < OUTC; ++c) { l[c] = __expf(l[c] - mx); se += l[c]; }
    const float inv = 1.f / se;
#pragma unroll
    for (int c = 0; c < OUTC; ++c) out[(size_t)m * OUTC + c] = l[c] * inv;
  }
}

extern "C" void kernel_launch(void* const* d_in, const int* in_sizes, int n_in,
                              void* d_out, int out_size, void* d_ws, size_t ws_size,
                              hipStream_t stream) {
  const float* x   = (const float*)d_in[0];
  const float* adj = (const float*)d_in[1];
  const float* W1  = (const float*)d_in[2];
  const float* b1  = (const float*)d_in[3];
  const float* W2  = (const float*)d_in[4];
  const float* b2  = (const float*)d_in[5];
  float* out = (float*)d_out;

  // Workspace layout (bf16): S1t [208][8192], H [8192][208], S2t [16][8192]
  __bf16* S1t = (__bf16*)d_ws;
  __bf16* H   = S1t + (size_t)HIDP * NNODE;
  __bf16* S2t = H   + (size_t)NNODE * HIDP;

  const size_t smem = (size_t)HIDP * KSTR * sizeof(__bf16);  // ~107 KB dynamic LDS
  k_gemm_xw1           <<<dim3(NNODE / 64), dim3(128), smem, stream>>>(x, W1, S1t);
  k_gemm_adj_s1        <<<dim3(NNODE / 64), dim3(128), smem, stream>>>(adj, S1t, b1, H);
  k_h_w2               <<<dim3(NNODE / 256), dim3(256), 0,   stream>>>(H, W2, S2t);
  k_gemm_adj_s2_softmax<<<dim3(NNODE / 64), dim3(128), 0,    stream>>>(adj, S2t, b2, out);
}
